// NerfRender_62500364091604
// MI455X (gfx1250) — compile-verified, matrix operand order checked by reading, not run on hardware
//
#include <hip/hip_runtime.h>
#include <hip/hip_bf16.h>

typedef __attribute__((ext_vector_type(16))) _Float16 v16h;
typedef __attribute__((ext_vector_type(8)))  float    v8f;

#define N_RAYS        65536
#define N_SAMP        128
#define RAYS_PER_BLK  16
#define NTHREADS      256
#define ELEMS         (RAYS_PER_BLK * N_SAMP)   /* 2048 */
#define EPT           (ELEMS / NTHREADS)        /* 8    */

union AFrag { v16h v; _Float16 e[16]; };
union CFrag { v8f  v; float    e[8];  };

__global__ __launch_bounds__(NTHREADS)
void nerf_render_kernel(const float* __restrict__ t,
                        const float* __restrict__ sigma,
                        const float* __restrict__ c,
                        float* __restrict__ outColor,   // [N,3]
                        float* __restrict__ outDepth,   // [N]
                        float* __restrict__ outWi)      // [N,128]
{
    __shared__ float ts[ELEMS];      // sorted t
    __shared__ float bufA[ELEMS];    // scan ping
    __shared__ float bufB[ELEMS];    // scan pong
    __shared__ float red[8 * 32];    // WMMA accumulator staging

    const int tid     = threadIdx.x;
    const int rayBase = blockIdx.x * RAYS_PER_BLK;
    const int gbase   = rayBase * N_SAMP;       // flat global element base

    // ---- 1. coalesced load of t into LDS ----
#pragma unroll
    for (int jj = 0; jj < EPT; ++jj) {
        int e = tid + jj * NTHREADS;
        ts[e] = t[gbase + e];
    }
    __syncthreads();

    // ---- 2. bitonic sort: 16 independent rows of 128, ascending ----
    for (int k = 2; k <= N_SAMP; k <<= 1) {
        for (int j = k >> 1; j >= 1; j >>= 1) {
#pragma unroll
            for (int m = 0; m < 4; ++m) {
                int q   = tid + m * NTHREADS;           // 1024 comparators
                int row = q >> 6;                       // 64 per row
                int p   = q & 63;
                int i   = ((p & ~(j - 1)) << 1) | (p & (j - 1));
                int l   = i | j;
                int bi  = row * N_SAMP + i;
                int bl  = row * N_SAMP + l;
                float a = ts[bi], b = ts[bl];
                bool up = ((i & k) == 0);
                if ((a > b) == up) { ts[bi] = b; ts[bl] = a; }
            }
            __syncthreads();
        }
    }

    // ---- 3. sdt = sigma(original order) * dt(sorted t), pad 0 at end ----
#pragma unroll
    for (int jj = 0; jj < EPT; ++jj) {
        int e = tid + jj * NTHREADS;
        int i = e & (N_SAMP - 1);
        float dt = (i < N_SAMP - 1) ? (ts[e + 1] - ts[e]) : 0.0f;
        bufA[e] = sigma[gbase + e] * dt;
    }
    __syncthreads();

    // ---- 4. inclusive scan per ray (Hillis-Steele, 7 steps) ----
    float* src = bufA;
    float* dst = bufB;
    for (int d = 1; d < N_SAMP; d <<= 1) {
#pragma unroll
        for (int jj = 0; jj < EPT; ++jj) {
            int e = tid + jj * NTHREADS;
            int i = e & (N_SAMP - 1);
            float v = src[e];
            if (i >= d) v += src[e - d];
            dst[e] = v;
        }
        __syncthreads();
        float* tmp = src; src = dst; dst = tmp;
    }
    const float* incl = src;   // final inclusive prefix sums
    float*       prod = dst;   // free buffer for WMMA products

    // ---- 5. weights wi = exp(-excl) - exp(-incl); stream out; cache c ----
    float wi[EPT], cr[EPT], cg[EPT], cb[EPT];
#pragma unroll
    for (int jj = 0; jj < EPT; ++jj) {
        int e = tid + jj * NTHREADS;
        int i = e & (N_SAMP - 1);
        float exI = __expf(-incl[e]);
        float exE = (i == 0) ? 1.0f : __expf(-incl[e - 1]);
        wi[jj] = exE - exI;
        outWi[gbase + e] = wi[jj];
        const float* cc = c + (size_t)(gbase + e) * 3;
        cr[jj] = cc[0]; cg[jj] = cc[1]; cb[jj] = cc[2];
    }

    // ---- 6. per-ray reductions via v_wmma_f32_16x16x32_f16 (B = ones) ----
    // q=0: depth = sum wi*t_sorted ; q=1..3: color RGB = sum wi*c
    for (int q = 0; q < 4; ++q) {
#pragma unroll
        for (int jj = 0; jj < EPT; ++jj) {
            int e = tid + jj * NTHREADS;
            float val;
            if (q == 0)      val = ts[e];
            else if (q == 1) val = cr[jj];
            else if (q == 2) val = cg[jj];
            else             val = cb[jj];
            prod[e] = wi[jj] * val;
        }
        __syncthreads();

        if (tid < 32) {                    // wave 0, EXEC all ones
            int ray = tid & 15;
            int h   = tid >> 4;            // K half select per A-matrix layout
            AFrag bf;
#pragma unroll
            for (int x = 0; x < 16; ++x) bf.e[x] = (_Float16)1.0f;
            CFrag acc;
#pragma unroll
            for (int x = 0; x < 8; ++x) acc.e[x] = 0.0f;
#pragma unroll
            for (int w = 0; w < 4; ++w) {  // 4 x K=32 covers 128 samples
                int kb = w * 32;
                AFrag a;
#pragma unroll
                for (int v = 0; v < 8; ++v) {
                    int koff = (v < 4) ? (2 * v + 8 * h)
                                       : (16 + 2 * (v - 4) + 8 * h);
                    a.e[2 * v]     = (_Float16)prod[ray * N_SAMP + kb + koff];
                    a.e[2 * v + 1] = (_Float16)prod[ray * N_SAMP + kb + koff + 1];
                }
                acc.v = __builtin_amdgcn_wmma_f32_16x16x32_f16(
                    false, a.v, false, bf.v, (short)0, acc.v, false, false);
            }
#pragma unroll
            for (int v = 0; v < 8; ++v) red[v * 32 + tid] = acc.e[v];
        }
        __syncthreads();

        if (tid < RAYS_PER_BLK) {
            int m = tid;
            // D layout: VGPR v holds M=v (lanes 0-15) and M=v+8 (lanes 16-31)
            float val = red[(m & 7) * 32 + ((m >> 3) ? 16 : 0)];
            int gray = rayBase + m;
            if (q == 0) outDepth[gray] = val;
            else        outColor[gray * 3 + (q - 1)] = val;
        }
        __syncthreads();   // protect prod/red reuse across q iterations
    }
}

extern "C" void kernel_launch(void* const* d_in, const int* in_sizes, int n_in,
                              void* d_out, int out_size, void* d_ws, size_t ws_size,
                              hipStream_t stream) {
    (void)in_sizes; (void)n_in; (void)out_size; (void)d_ws; (void)ws_size;
    const float* t     = (const float*)d_in[0];
    const float* sigma = (const float*)d_in[1];
    const float* c     = (const float*)d_in[2];
    float* out      = (float*)d_out;
    float* outColor = out;                       // [65536,3]
    float* outDepth = out + (size_t)N_RAYS * 3;  // [65536]
    float* outWi    = out + (size_t)N_RAYS * 4;  // [65536,128]
    dim3 grid(N_RAYS / RAYS_PER_BLK);
    nerf_render_kernel<<<grid, NTHREADS, 0, stream>>>(t, sigma, c,
                                                      outColor, outDepth, outWi);
}
